// CustomLSTM_73418170958323
// MI455X (gfx1250) — matrix operand order repaired and checked by reading, not compiled, not run on hardware
//
#include <hip/hip_runtime.h>
#include <hip/hip_bf16.h>
#include <math.h>

// Problem dims
#define SEQ   8192
#define VOCAB 256
#define HID   2048
#define OUTD  256
#define ZN    (4 * HID)      // fused gate width (i,f,g,o) = 8192
#define KT    (HID / 32)     // 64 K-chunks of 32 for 16x16x32 WMMA
#define NWG   64             // workgroups in recurrence
#define WPB   8              // waves per block (wave32)
#define TPB   256

typedef __attribute__((ext_vector_type(16))) __bf16 v16bf;
typedef __attribute__((ext_vector_type(8)))  float  v8f;

struct __align__(16) B128 { unsigned w[4]; };

union TileU {
  v16bf v;
  unsigned short u[16];
  B128 q[2];
};

__device__ inline unsigned short f2bf(float f) {
  union { float f; unsigned u; } c; c.f = f;
  unsigned r = c.u + 0x7FFFu + ((c.u >> 16) & 1u);  // round-to-nearest-even
  return (unsigned short)(r >> 16);
}
__device__ inline float sigf(float z) { return 1.0f / (1.0f + __expf(-z)); }

// ---------------------------------------------------------------------------
// init: zero bf16 h state and the grid-barrier counter (each launch -> determ.)
// ---------------------------------------------------------------------------
__global__ void __launch_bounds__(TPB) init_state(unsigned short* hbuf,
                                                  unsigned* cnt) {
  const int tid = threadIdx.x;
  for (int i = tid; i < HID; i += TPB) hbuf[i] = 0;
  if (tid == 0) cnt[0] = 0u;
}

// ---------------------------------------------------------------------------
// EW[g][v][j] = emb[v,:] . W_g[:,j] + b_g[j] + c_g[j]   (tiny: 256-deep dots)
// ---------------------------------------------------------------------------
__global__ void __launch_bounds__(TPB) ew_kernel(
    const float* __restrict__ emb,
    const float* __restrict__ W0, const float* __restrict__ W1,
    const float* __restrict__ W2, const float* __restrict__ W3,
    const float* __restrict__ b0, const float* __restrict__ b1,
    const float* __restrict__ b2, const float* __restrict__ b3,
    const float* __restrict__ c0, const float* __restrict__ c1,
    const float* __restrict__ c2, const float* __restrict__ c3,
    float* __restrict__ EW) {
  const int idx = blockIdx.x * TPB + threadIdx.x;    // [0, 4*256*2048)
  const int g = idx >> 19;
  const int r = idx & ((VOCAB * HID) - 1);
  const int v = r >> 11;
  const int j = r & (HID - 1);
  const float* W = (g == 0) ? W0 : (g == 1) ? W1 : (g == 2) ? W2 : W3;
  const float* B = (g == 0) ? b0 : (g == 1) ? b1 : (g == 2) ? b2 : b3;
  const float* C = (g == 0) ? c0 : (g == 1) ? c1 : (g == 2) ? c2 : c3;
  float acc = B[j] + C[j];
  const float* er = emb + (size_t)v * VOCAB;
  for (int k = 0; k < VOCAB; ++k)
    acc = fmaf(er[k], W[(size_t)k * HID + j], acc);
  EW[idx] = acc;
}

// ---------------------------------------------------------------------------
// Pack fused U (2048 x 8192, gate-major in N) into per-tile WMMA-B bf16
// layout: tile (tn,tk) is 1024B; lane l owns 32B: VGPR v half hh holds
// element (k = tk*32 + (l>=16)*16 + 2v+hh, n = tn*16 + (l&15)).
// ---------------------------------------------------------------------------
__global__ void __launch_bounds__(TPB) pack_u(
    const float* __restrict__ U0, const float* __restrict__ U1,
    const float* __restrict__ U2, const float* __restrict__ U3,
    unsigned short* __restrict__ Up) {
  const int idx = blockIdx.x * TPB + threadIdx.x;    // [0, 2048*8192)
  const int tile = idx >> 9;
  const int r = idx & 511;
  const int l = r >> 4;
  const int q = r & 15;                              // = 2v + hh = k offset
  const int tn = tile >> 6;                          // / KT
  const int tk = tile & (KT - 1);
  const int k = tk * 32 + (l >> 4) * 16 + q;
  const int n = tn * 16 + (l & 15);
  const int g = n >> 11;
  const int j = n & (HID - 1);
  const float* U = (g == 0) ? U0 : (g == 1) ? U1 : (g == 2) ? U2 : U3;
  Up[idx] = f2bf(U[(size_t)k * HID + j]);
}

// Same packing for V_w (2048 x 256)
__global__ void __launch_bounds__(TPB) pack_v(const float* __restrict__ Vw,
                                              unsigned short* __restrict__ Vp) {
  const int idx = blockIdx.x * TPB + threadIdx.x;    // [0, 2048*256)
  const int tile = idx >> 9;
  const int r = idx & 511;
  const int l = r >> 4;
  const int q = r & 15;
  const int tn = tile >> 6;
  const int tk = tile & (KT - 1);
  const int k = tk * 32 + (l >> 4) * 16 + q;
  const int n = tn * 16 + (l & 15);
  Vp[idx] = f2bf(Vw[(size_t)k * OUTD + n]);
}

// ---------------------------------------------------------------------------
// Persistent recurrence. WG b owns h-columns [32b, 32b+32):
//   wave w (0..7): gate g = w>>1, 16-col tile nt = 2b + (w&1) of that gate.
// Gate exchange is intra-WG (LDS); only ONE grid barrier per step (h bcast).
// c-state lives in registers of the 32 owner threads. A-tile rows 1..15 are
// DON'T-CARE (only D row 0 is read) -> uniform, unmasked lane-group loads.
// ---------------------------------------------------------------------------
__global__ void __launch_bounds__(TPB, 1) lstm_recur(
    const int* __restrict__ x,
    const float* __restrict__ EW,
    const unsigned short* __restrict__ Upack,
    unsigned short* __restrict__ hs,
    unsigned short* hbuf,                 // h as bf16, global broadcast
    unsigned* cnt, float* out_tail) {
  __shared__ unsigned short hlds[HID];    // h as bf16, 4 KB
  __shared__ float zlds[4 * 32];          // activated gates for our 32 cols
  const int tid  = threadIdx.x;
  const int lane = tid & 31;
  const int w    = tid >> 5;              // wave in WG: 0..7
  const int g    = w >> 1;                // gate of this wave
  const int nt   = blockIdx.x * 2 + (w & 1);  // 16-col tile within gate
  const int jb   = nt * 16;               // h-column base of this tile
  const unsigned short* tptr =
      Upack + (size_t)(g * (HID / 16) + nt) * KT * 512;
  const int abase = (lane >> 4) << 4;     // lane group 0: +0, group 1: +16

  float creg = 0.f;                       // c[j] register state (tid < 32)
  const int j = blockIdx.x * 32 + tid;    // h column owned in update phase

  for (int t = 0; t < SEQ; ++t) {
    // stage h: bf16 global -> bf16 LDS, 16B per thread (256 x 8 u16 = 2048)
    {
      const int i = tid * 8;
      *(B128*)(hlds + i) = *(const B128*)(hbuf + i);
    }
    __syncthreads();

    // matvec: 4 independent accumulator chains over the 64 K-chunks.
    // Keep the loop rolled: the 4-wide body is the intended pipeline depth;
    // further unrolling spills B tiles to scratch.
    v8f acc0 = {}, acc1 = {}, acc2 = {}, acc3 = {};
    #pragma clang loop unroll(disable)
    for (int kc = 0; kc < KT; kc += 4) {
      TileU a0, a1, a2, a3, b0, b1, b2, b3;
      const B128* h0 = (const B128*)(hlds + (kc + 0) * 32 + abase);
      const B128* h1 = (const B128*)(hlds + (kc + 1) * 32 + abase);
      const B128* h2 = (const B128*)(hlds + (kc + 2) * 32 + abase);
      const B128* h3 = (const B128*)(hlds + (kc + 3) * 32 + abase);
      a0.q[0] = h0[0]; a0.q[1] = h0[1];
      a1.q[0] = h1[0]; a1.q[1] = h1[1];
      a2.q[0] = h2[0]; a2.q[1] = h2[1];
      a3.q[0] = h3[0]; a3.q[1] = h3[1];
      const B128* t0 = (const B128*)(tptr + (size_t)(kc + 0) * 512);
      const B128* t1 = (const B128*)(tptr + (size_t)(kc + 1) * 512);
      const B128* t2 = (const B128*)(tptr + (size_t)(kc + 2) * 512);
      const B128* t3 = (const B128*)(tptr + (size_t)(kc + 3) * 512);
      b0.q[0] = t0[lane * 2]; b0.q[1] = t0[lane * 2 + 1];
      b1.q[0] = t1[lane * 2]; b1.q[1] = t1[lane * 2 + 1];
      b2.q[0] = t2[lane * 2]; b2.q[1] = t2[lane * 2 + 1];
      b3.q[0] = t3[lane * 2]; b3.q[1] = t3[lane * 2 + 1];
      if (kc + 4 < KT)
        __builtin_prefetch(tptr + (size_t)(kc + 4) * 512 + lane * 16, 0, 1);
      acc0 = __builtin_amdgcn_wmma_f32_16x16x32_bf16(false, a0.v, false, b0.v,
                                                     (short)0, acc0, false, false);
      acc1 = __builtin_amdgcn_wmma_f32_16x16x32_bf16(false, a1.v, false, b1.v,
                                                     (short)0, acc1, false, false);
      acc2 = __builtin_amdgcn_wmma_f32_16x16x32_bf16(false, a2.v, false, b2.v,
                                                     (short)0, acc2, false, false);
      acc3 = __builtin_amdgcn_wmma_f32_16x16x32_bf16(false, a3.v, false, b3.v,
                                                     (short)0, acc3, false, false);
    }

    if (lane < 16) {                      // D row 0: lanes 0..15, VGPR 0
      const int xt = x[t];
      float z = (acc0[0] + acc1[0]) + (acc2[0] + acc3[0]);
      z += EW[((size_t)g * VOCAB + xt) * HID + jb + lane];
      zlds[g * 32 + (w & 1) * 16 + lane] = (g == 2) ? tanhf(z) : sigf(z);
    }
    __syncthreads();

    // c/h update for our 32 columns (one wave)
    if (tid < 32) {
      const float iv = zlds[tid];
      const float fv = zlds[32 + tid];
      const float gv = zlds[64 + tid];
      const float ov = zlds[96 + tid];
      const float cn = fv * creg + iv * gv;
      const float hn = ov * tanhf(cn);
      creg = cn;
      const unsigned short hb = f2bf(hn);
      hbuf[j] = hb;
      hs[(size_t)t * HID + j] = hb;
      if (t == SEQ - 1) { out_tail[j] = hn; out_tail[HID + j] = cn; }
    }

    // single grid barrier per step: h fully published before restaging
    __syncthreads();
    if (tid == 0) {
      __hip_atomic_fetch_add(cnt, 1u, __ATOMIC_RELEASE,
                             __HIP_MEMORY_SCOPE_AGENT);
      const unsigned tgt = (unsigned)(t + 1) * NWG;
      while (__hip_atomic_load(cnt, __ATOMIC_ACQUIRE,
                               __HIP_MEMORY_SCOPE_AGENT) < tgt)
        __builtin_amdgcn_s_sleep(1);
    }
    __syncthreads();
    __threadfence();                      // invalidate stale h lines (all CUs)
  }
}

// ---------------------------------------------------------------------------
// out = hs @ V_w + V_b : one 16x16 output tile per wave, K = 2048 in 64 chunks
// ---------------------------------------------------------------------------
__global__ void __launch_bounds__(TPB, 1) out_gemm(
    const unsigned short* __restrict__ hs,
    const unsigned short* __restrict__ Vpack,
    const float* __restrict__ Vb,
    float* __restrict__ out) {
  const int tid  = threadIdx.x;
  const int lane = tid & 31;
  const int wt   = blockIdx.x * WPB + (tid >> 5);    // [0, 8192) tiles
  const int tr   = wt >> 4;                          // row tile  [0,512)
  const int tc   = wt & 15;                          // col tile  [0,16)
  const int m    = lane & 15;
  const int hi   = lane >> 4;

  v8f acc0 = {}, acc1 = {};
  const unsigned short* arow = hs + (size_t)(tr * 16 + m) * HID + hi * 16;
  const unsigned short* tptr = Vpack + (size_t)tc * KT * 512;
  #pragma clang loop unroll(disable)
  for (int kc = 0; kc < KT; kc += 2) {
    TileU a0, a1, b0, b1;
    const B128* aq0 = (const B128*)(arow + (kc + 0) * 32);
    const B128* aq1 = (const B128*)(arow + (kc + 1) * 32);
    a0.q[0] = aq0[0]; a0.q[1] = aq0[1];
    a1.q[0] = aq1[0]; a1.q[1] = aq1[1];
    const B128* t0 = (const B128*)(tptr + (size_t)(kc + 0) * 512);
    const B128* t1 = (const B128*)(tptr + (size_t)(kc + 1) * 512);
    b0.q[0] = t0[lane * 2]; b0.q[1] = t0[lane * 2 + 1];
    b1.q[0] = t1[lane * 2]; b1.q[1] = t1[lane * 2 + 1];
    if (kc + 2 < KT)
      __builtin_prefetch(arow + (kc + 2) * 32, 0, 1);
    acc0 = __builtin_amdgcn_wmma_f32_16x16x32_bf16(false, a0.v, false, b0.v,
                                                   (short)0, acc0, false, false);
    acc1 = __builtin_amdgcn_wmma_f32_16x16x32_bf16(false, a1.v, false, b1.v,
                                                   (short)0, acc1, false, false);
  }
  const int col = tc * 16 + m;
  const float bias = Vb[col];
  #pragma unroll
  for (int r = 0; r < 8; ++r) {
    const int row = tr * 16 + r + hi * 8;            // C layout: M = r + 8*hi
    out[(size_t)row * OUTD + col] = (acc0[r] + acc1[r]) + bias;
  }
}

// ---------------------------------------------------------------------------
extern "C" void kernel_launch(void* const* d_in, const int* in_sizes, int n_in,
                              void* d_out, int out_size, void* d_ws,
                              size_t ws_size, hipStream_t stream) {
  (void)in_sizes; (void)n_in; (void)out_size; (void)ws_size;
  const int*   x   = (const int*)  d_in[0];
  const float* emb = (const float*)d_in[1];
  const float* Wi  = (const float*)d_in[2],  *bi = (const float*)d_in[3];
  const float* Ui  = (const float*)d_in[4],  *ci = (const float*)d_in[5];
  const float* Wf  = (const float*)d_in[6],  *bf = (const float*)d_in[7];
  const float* Uf  = (const float*)d_in[8],  *cf = (const float*)d_in[9];
  const float* Wg  = (const float*)d_in[10], *bg = (const float*)d_in[11];
  const float* Ug  = (const float*)d_in[12], *cg = (const float*)d_in[13];
  const float* Wo  = (const float*)d_in[14], *bo = (const float*)d_in[15];
  const float* Uo  = (const float*)d_in[16], *co = (const float*)d_in[17];
  const float* Vw  = (const float*)d_in[18], *Vb = (const float*)d_in[19];

  char* ws = (char*)d_ws;
  size_t off = 0;
  float*          EW  = (float*)(ws + off);          off += (size_t)4 * VOCAB * HID * 4;  // 8 MB
  unsigned short* Up  = (unsigned short*)(ws + off); off += (size_t)HID * ZN * 2;          // 32 MB
  unsigned short* Vp  = (unsigned short*)(ws + off); off += (size_t)HID * OUTD * 2;        // 1 MB
  unsigned short* hsb = (unsigned short*)(ws + off); off += (size_t)SEQ * HID * 2;         // 32 MB
  unsigned short* hbuf = (unsigned short*)(ws + off); off += (size_t)HID * 2;
  unsigned*       cnt  = (unsigned*)(ws + off);      off += 256;

  float* out      = (float*)d_out;
  float* out_tail = out + (size_t)SEQ * OUTD;        // h_T then c_T

  init_state<<<1, TPB, 0, stream>>>(hbuf, cnt);
  ew_kernel<<<(4 * VOCAB * HID) / TPB, TPB, 0, stream>>>(
      emb, Wi, Wf, Wg, Wo, bi, bf, bg, bo, ci, cf, cg, co, EW);
  pack_u<<<((size_t)HID * ZN) / TPB, TPB, 0, stream>>>(Ui, Uf, Ug, Uo, Up);
  pack_v<<<((size_t)HID * OUTD) / TPB, TPB, 0, stream>>>(Vw, Vp);
  lstm_recur<<<NWG, TPB, 0, stream>>>(x, EW, Up, hsb, hbuf, cnt, out_tail);
  out_gemm<<<(SEQ / 16) * (OUTD / 16) / WPB, TPB, 0, stream>>>(hsb, Vp, Vb, out);
}
